// SoftAttentionMax_45578192945931
// MI455X (gfx1250) — matrix operand order repaired and checked by reading, not compiled, not run on hardware
//
#include <hip/hip_runtime.h>
#include <hip/hip_bf16.h>

// SAGAN non-local attention block for MI455X (gfx1250, wave32, WMMA bf16).
// B=8, H=W=64, C=512, cfg=64, ch=256. All GEMMs via v_wmma_f32_16x16x32_bf16.

typedef __attribute__((ext_vector_type(16))) __bf16 v16bf;
typedef __attribute__((ext_vector_type(8)))  __bf16 v8bf;
typedef __attribute__((ext_vector_type(4)))  __bf16 v4bf;
typedef __attribute__((ext_vector_type(8)))  float  v8f;

#define DEV __device__ __forceinline__

static constexpr int BB  = 8;
static constexpr int HH  = 64;
static constexpr int WW  = 64;
static constexpr int CC  = 512;
static constexpr int CFG = 64;
static constexpr int CH  = 256;
static constexpr int HW  = HH * WW;          // 4096
static constexpr int HWD = (HH/2) * (WW/2);  // 1024
static constexpr int NPIX = BB * HW;         // 32768

DEV v8f wmma_bf16(v16bf a, v16bf b, v8f c) {
  return __builtin_amdgcn_wmma_f32_16x16x32_bf16(
      /*neg_a=*/false, a, /*neg_b=*/false, b,
      /*c_mod=*/(short)0, c, /*reuse_a=*/false, /*reuse_b=*/false);
}

// A fragment (16x32 bf16, M=rows): lane<16 holds row l16, K = k0+0..15;
// lane>=16 holds row l16, K = k0+16..31.
DEV v16bf load_A_bf16(const __bf16* rowp, int k0, int hf) {
  return *(const v16bf*)(rowp + k0 + 16 * hf);        // 32B aligned
}

// B fragment (32x16 bf16): lane<16 holds column n=l16, K=k0+0..15; lane>=16 column
// n=l16, K=k0+16..31. Source is B^T row-major (row n, contiguous K) -> one 32B load.
DEV v16bf load_B_bf16(const __bf16* rowTp, int k0, int hf) {
  return *(const v16bf*)(rowTp + k0 + 16 * hf);       // 32B aligned
}

// ----------------------------------------------------------------- x -> bf16
__global__ void sam_cvt_x(const float* __restrict__ src, __bf16* __restrict__ dst) {
  size_t idx = ((size_t)blockIdx.x * blockDim.x + threadIdx.x) * 4;
  if (idx >= (size_t)NPIX * CC) return;
  float4 v = *(const float4*)(src + idx);
  v4bf o;
  o[0] = (__bf16)v.x; o[1] = (__bf16)v.y; o[2] = (__bf16)v.z; o[3] = (__bf16)v.w;
  *(v4bf*)(dst + idx) = o;
}

// ---------------------------------------------------------------- weight prep
__global__ void sam_transpose_w(const float* __restrict__ src, __bf16* __restrict__ dst,
                                int R, int S) {
  int idx = blockIdx.x * blockDim.x + threadIdx.x;
  if (idx >= R * S) return;
  int r = idx / S, s = idx - r * S;
  dst[(size_t)s * R + r] = (__bf16)src[idx];          // dst[S][R] = src^T
}

// -------------------------------------------------------- projection GEMM
// Out[N x F] = X[N x C] * W[C x F] + bias  (WT = W^T bf16 [F][C]).
// Each wave: one 16x64 output strip (4 N-tiles) -> A fragment reused 4x.
__global__ void sam_proj_gemm(const __bf16* __restrict__ Xbf, const __bf16* __restrict__ WT,
                              const float* __restrict__ bias, __bf16* __restrict__ Out,
                              int C, int F) {
  int tM = blockIdx.x, nG = blockIdx.y;               // nG covers 64 columns
  int lane = threadIdx.x & 31, hf = lane >> 4, l16 = lane & 15;
  const __bf16* arow = Xbf + (size_t)(tM * 16 + l16) * C;
  const __bf16* brow[4];
#pragma unroll
  for (int t = 0; t < 4; ++t)
    brow[t] = WT + (size_t)((nG * 4 + t) * 16 + l16) * C;
  v8f acc[4] = {{}, {}, {}, {}};
  for (int k0 = 0; k0 < C; k0 += 32) {
    v16bf a = load_A_bf16(arow, k0, hf);
#pragma unroll
    for (int t = 0; t < 4; ++t)
      acc[t] = wmma_bf16(a, load_B_bf16(brow[t], k0, hf), acc[t]);
  }
#pragma unroll
  for (int t = 0; t < 4; ++t) {
    int col = (nG * 4 + t) * 16 + l16;
    float bv = bias[col];
#pragma unroll
    for (int v = 0; v < 8; ++v) {
      size_t row = (size_t)tM * 16 + v + 8 * hf;
      Out[row * F + col] = (__bf16)(acc[t][v] + bv);
    }
  }
}

// ------------------------------------------------------------------ pooling
__global__ void sam_pool_phi(const __bf16* __restrict__ full, __bf16* __restrict__ out) {
  int idx = blockIdx.x * blockDim.x + threadIdx.x;            // B*HWD*CFG
  if (idx >= BB * HWD * CFG) return;
  int f = idx & (CFG - 1);
  int i = (idx >> 6) & (HWD - 1);
  int b = idx >> 16;
  int ph = i >> 5, pw = i & 31;
  size_t base = (size_t)b * HW;
  auto val = [&](int h, int w) {
    return (float)full[(base + (size_t)h * WW + w) * CFG + f];
  };
  float m = fmaxf(fmaxf(val(2*ph, 2*pw), val(2*ph, 2*pw+1)),
                  fmaxf(val(2*ph+1, 2*pw), val(2*ph+1, 2*pw+1)));
  out[((size_t)b * HWD + i) * CFG + f] = (__bf16)m;
}

__global__ void sam_pool_gT(const __bf16* __restrict__ full, __bf16* __restrict__ outT) {
  int idx = blockIdx.x * blockDim.x + threadIdx.x;            // B*CH*HWD
  if (idx >= BB * CH * HWD) return;
  int i = idx & (HWD - 1);
  int c = (idx >> 10) & (CH - 1);
  int b = idx >> 18;
  int ph = i >> 5, pw = i & 31;
  size_t base = (size_t)b * HW;
  auto val = [&](int h, int w) {
    return (float)full[(base + (size_t)h * WW + w) * CH + c];
  };
  float m = fmaxf(fmaxf(val(2*ph, 2*pw), val(2*ph, 2*pw+1)),
                  fmaxf(val(2*ph+1, 2*pw), val(2*ph+1, 2*pw+1)));
  outT[((size_t)b * CH + c) * HWD + i] = (__bf16)m;           // transposed [b][c][i]
}

// -------------------------------------------- pass 1: row softmax stats (m, l)
// s[b,i,j] = phi[b,i,:] . theta[b,j,:]; softmax over j. One block per (iTile, b),
// 4 waves split the 256 j-tiles, online-softmax merge across lanes then waves.
__global__ void sam_softmax_stats(const __bf16* __restrict__ phiP,
                                  const __bf16* __restrict__ theta,
                                  float* __restrict__ mOut, float* __restrict__ lOut) {
  int iTile = blockIdx.x, b = blockIdx.y;
  int wave = threadIdx.x >> 5, lane = threadIdx.x & 31;
  int hf = lane >> 4, l16 = lane & 15;
  const __bf16* phiRow = phiP + ((size_t)b * HWD + iTile * 16 + l16) * CFG;
  const __bf16* thB    = theta + (size_t)b * HW * CFG;
  float m[8], l[8];
#pragma unroll
  for (int v = 0; v < 8; ++v) { m[v] = -__builtin_inff(); l[v] = 0.f; }
  v16bf a0 = load_A_bf16(phiRow, 0,  hf);
  v16bf a1 = load_A_bf16(phiRow, 32, hf);
  for (int jt = wave; jt < HW / 16; jt += 4) {
    const __bf16* thRow = thB + (size_t)(jt * 16 + l16) * CFG;
    v8f s = {};
    s = wmma_bf16(a0, load_B_bf16(thRow, 0,  hf), s);
    s = wmma_bf16(a1, load_B_bf16(thRow, 32, hf), s);
#pragma unroll
    for (int v = 0; v < 8; ++v) {
      float sv = s[v];
      float mn = fmaxf(m[v], sv);
      l[v] = l[v] * __expf(m[v] - mn) + __expf(sv - mn);
      m[v] = mn;
    }
  }
  // reduce (m,l) across the 16 lanes of each half (xor masks stay within half)
#pragma unroll
  for (int off = 1; off < 16; off <<= 1) {
#pragma unroll
    for (int v = 0; v < 8; ++v) {
      float mo = __shfl_xor(m[v], off, 32);
      float lo = __shfl_xor(l[v], off, 32);
      float mn = fmaxf(m[v], mo);
      l[v] = l[v] * __expf(m[v] - mn) + lo * __expf(mo - mn);
      m[v] = mn;
    }
  }
  __shared__ float mSh[4][16], lSh[4][16];
  if (l16 == 0) {
#pragma unroll
    for (int v = 0; v < 8; ++v) { mSh[wave][v + 8*hf] = m[v]; lSh[wave][v + 8*hf] = l[v]; }
  }
  __syncthreads();
  if (threadIdx.x < 16) {
    int r = threadIdx.x;
    float mm = mSh[0][r], ll = lSh[0][r];
#pragma unroll
    for (int w = 1; w < 4; ++w) {
      float mo = mSh[w][r], lo = lSh[w][r];
      float mn = fmaxf(mm, mo);
      ll = ll * __expf(mm - mn) + lo * __expf(mo - mn);
      mm = mn;
    }
    size_t o = (size_t)b * HWD + iTile * 16 + r;
    mOut[o] = mm; lOut[o] = ll;
  }
}

// ------------------- pass 2: o[b,j,c] = sum_i exp(s[i,j]-m_i)/l_i * g[i,c]
// One block per (jTile, b); 4 waves x 4 c-tiles each cover all 256 channels.
__global__ void sam_attention(const __bf16* __restrict__ theta,
                              const __bf16* __restrict__ phiP,
                              const __bf16* __restrict__ gTP,
                              const float* __restrict__ mBuf,
                              const float* __restrict__ lBuf,
                              __bf16* __restrict__ oBuf) {
  int jTile = blockIdx.x, b = blockIdx.y;
  int wave = threadIdx.x >> 5, lane = threadIdx.x & 31;
  int hf = lane >> 4, l16 = lane & 15;
  __shared__ __bf16 pT[4][16][32];                     // per-wave p tile [j][i]
  const __bf16* thRow = theta + ((size_t)b * HW + jTile * 16 + l16) * CFG;
  const __bf16* phiB  = phiP + (size_t)b * HWD * CFG;
  const __bf16* gB    = gTP  + (size_t)b * CH * HWD;
  const float*  mB    = mBuf + (size_t)b * HWD;
  const float*  lB    = lBuf + (size_t)b * HWD;
  int cBase = wave * 64;
  v16bf ta0 = load_A_bf16(thRow, 0,  hf);
  v16bf ta1 = load_A_bf16(thRow, 32, hf);
  v8f acc[4] = {{}, {}, {}, {}};
  for (int i0 = 0; i0 < HWD; i0 += 32) {
    // sT[j,i] tiles + normalized exp -> LDS (bf16)
#pragma unroll
    for (int it = 0; it < 2; ++it) {
      const __bf16* phiRow = phiB + (size_t)(i0 + it * 16 + l16) * CFG;
      v8f s = {};
      s = wmma_bf16(ta0, load_B_bf16(phiRow, 0,  hf), s);
      s = wmma_bf16(ta1, load_B_bf16(phiRow, 32, hf), s);
      int idx = i0 + it * 16 + l16;                    // i index held by this lane
      float mi = mB[idx];
      float inv = 1.0f / lB[idx];
#pragma unroll
      for (int v = 0; v < 8; ++v)
        pT[wave][v + 8 * hf][it * 16 + l16] = (__bf16)(__expf(s[v] - mi) * inv);
    }
    asm volatile("s_wait_dscnt 0" ::: "memory");       // LDS RAW within wave
    v16bf a = load_A_bf16(&pT[wave][l16][0], 0, hf);   // p as A fragment [16j x 32i]
#pragma unroll
    for (int ct = 0; ct < 4; ++ct) {
      const __bf16* gRow = gB + (size_t)(cBase + ct * 16 + l16) * HWD + i0;
      acc[ct] = wmma_bf16(a, load_B_bf16(gRow, 0, hf), acc[ct]);
    }
  }
#pragma unroll
  for (int ct = 0; ct < 4; ++ct)
#pragma unroll
    for (int v = 0; v < 8; ++v) {
      size_t row = (size_t)b * HW + jTile * 16 + v + 8 * hf;
      oBuf[row * CH + cBase + ct * 16 + l16] = (__bf16)acc[ct][v];
    }
}

// ---------------------- final: out = gamma * (o @ o3_w + b) + x   (f32 out)
// Each wave: 16x64 output strip (4 N-tiles), A fragment reused 4x.
__global__ void sam_output_gemm(const __bf16* __restrict__ oBuf,
                                const __bf16* __restrict__ o3T,
                                const float* __restrict__ o3b,
                                const float* __restrict__ gamma,
                                const float* __restrict__ x,
                                float* __restrict__ out) {
  int tM = blockIdx.x, nG = blockIdx.y;               // nG covers 64 of 512 cols
  int lane = threadIdx.x & 31, hf = lane >> 4, l16 = lane & 15;
  const __bf16* arow = oBuf + (size_t)(tM * 16 + l16) * CH;
  const __bf16* brow[4];
#pragma unroll
  for (int t = 0; t < 4; ++t)
    brow[t] = o3T + (size_t)((nG * 4 + t) * 16 + l16) * CH;
  v8f acc[4] = {{}, {}, {}, {}};
  for (int k0 = 0; k0 < CH; k0 += 32) {
    v16bf a = load_A_bf16(arow, k0, hf);
#pragma unroll
    for (int t = 0; t < 4; ++t)
      acc[t] = wmma_bf16(a, load_B_bf16(brow[t], k0, hf), acc[t]);
  }
  float g = gamma[0];
#pragma unroll
  for (int t = 0; t < 4; ++t) {
    int cc = (nG * 4 + t) * 16 + l16;
    float bv = o3b[cc];
#pragma unroll
    for (int v = 0; v < 8; ++v) {
      size_t n = (size_t)tM * 16 + v + 8 * hf;
      size_t idx = n * CC + cc;
      out[idx] = g * (acc[t][v] + bv) + x[idx];
    }
  }
}

// --------------------------------------------------------------------- host
extern "C" void kernel_launch(void* const* d_in, const int* in_sizes, int n_in,
                              void* d_out, int out_size, void* d_ws, size_t ws_size,
                              hipStream_t stream) {
  (void)in_sizes; (void)n_in; (void)out_size; (void)ws_size;
  const float* x       = (const float*)d_in[0];
  const float* theta_w = (const float*)d_in[1];
  const float* theta_b = (const float*)d_in[2];
  const float* phi_w   = (const float*)d_in[3];
  const float* phi_b   = (const float*)d_in[4];
  const float* g_w     = (const float*)d_in[5];
  const float* g_b     = (const float*)d_in[6];
  const float* o3_w    = (const float*)d_in[7];
  const float* o3_b    = (const float*)d_in[8];
  const float* gamma   = (const float*)d_in[9];
  float* out = (float*)d_out;

  char* w = (char*)d_ws;
  size_t off = 0;
  auto alloc = [&](size_t bytes) {
    void* p = w + off;
    off = (off + bytes + 255) & ~(size_t)255;
    return p;
  };
  __bf16* xbf    = (__bf16*)alloc((size_t)NPIX * CC * 2);       // 32 MB
  __bf16* thetaT = (__bf16*)alloc((size_t)CFG * CC * 2);        // [64][512]
  __bf16* phiT   = (__bf16*)alloc((size_t)CFG * CC * 2);        // [64][512]
  __bf16* gTw    = (__bf16*)alloc((size_t)CH  * CC * 2);        // [256][512]
  __bf16* o3T    = (__bf16*)alloc((size_t)CC  * CH * 2);        // [512][256]
  __bf16* theta  = (__bf16*)alloc((size_t)NPIX * CFG * 2);      // 4 MB
  __bf16* phiF   = (__bf16*)alloc((size_t)NPIX * CFG * 2);      // 4 MB
  __bf16* gF     = (__bf16*)alloc((size_t)NPIX * CH * 2);       // 16 MB
  __bf16* phiP   = (__bf16*)alloc((size_t)BB * HWD * CFG * 2);  // 1 MB
  __bf16* gTP    = (__bf16*)alloc((size_t)BB * CH * HWD * 2);   // 4 MB
  float*  mBuf   = (float*)alloc((size_t)BB * HWD * 4);
  float*  lBuf   = (float*)alloc((size_t)BB * HWD * 4);
  __bf16* oBuf   = (__bf16*)alloc((size_t)NPIX * CH * 2);       // 16 MB
  // total workspace ~= 80 MB

  // 0) x -> bf16 (once; avoids re-converting x per GEMM column tile)
  sam_cvt_x<<<(NPIX * CC / 4 + 255) / 256, 256, 0, stream>>>(x, xbf);

  // 1) weights -> bf16 transposed
  sam_transpose_w<<<(CC*CFG + 255)/256, 256, 0, stream>>>(theta_w, thetaT, CC, CFG);
  sam_transpose_w<<<(CC*CFG + 255)/256, 256, 0, stream>>>(phi_w,   phiT,   CC, CFG);
  sam_transpose_w<<<(CC*CH  + 255)/256, 256, 0, stream>>>(g_w,     gTw,    CC, CH);
  sam_transpose_w<<<(CH*CC  + 255)/256, 256, 0, stream>>>(o3_w,    o3T,    CH, CC);

  // 2) projections (WMMA bf16, 16x64 strip per wave)
  sam_proj_gemm<<<dim3(NPIX/16, CFG/64), 32, 0, stream>>>(xbf, thetaT, theta_b, theta, CC, CFG);
  sam_proj_gemm<<<dim3(NPIX/16, CFG/64), 32, 0, stream>>>(xbf, phiT,   phi_b,   phiF,  CC, CFG);
  sam_proj_gemm<<<dim3(NPIX/16, CH /64), 32, 0, stream>>>(xbf, gTw,    g_b,     gF,    CC, CH);

  // 3) 2x2 maxpool (phi normal layout, g transposed for B-operand contiguity)
  sam_pool_phi<<<(BB*HWD*CFG + 255)/256, 256, 0, stream>>>(phiF, phiP);
  sam_pool_gT <<<(BB*CH*HWD  + 255)/256, 256, 0, stream>>>(gF, gTP);

  // 4) softmax stats (flash pass 1)
  sam_softmax_stats<<<dim3(HWD/16, BB), 128, 0, stream>>>(phiP, theta, mBuf, lBuf);

  // 5) attention o = beta^T @ g (flash pass 2)
  sam_attention<<<dim3(HW/16, BB), 128, 0, stream>>>(theta, phiP, gTP, mBuf, lBuf, oBuf);

  // 6) final projection + gamma + residual
  sam_output_gemm<<<dim3(NPIX/16, CC/16/4), 32, 0, stream>>>(oBuf, o3T, o3_b, gamma, x, out);
}
